// SpatialContextEncoder_25391846654796
// MI455X (gfx1250) — matrix operand order repaired and checked by reading, not compiled, not run on hardware
//
#include <hip/hip_runtime.h>
#include <hip/hip_bf16.h>
#include <stdint.h>

typedef __attribute__((ext_vector_type(16))) __bf16        bf16x16;
typedef __attribute__((ext_vector_type(8)))  float         f32x8;
typedef __attribute__((ext_vector_type(4)))  unsigned int  u32x4;
typedef __attribute__((ext_vector_type(2)))  unsigned int  u32x2;
typedef __attribute__((ext_vector_type(8)))  int           i32x8;
typedef __attribute__((ext_vector_type(4)))  int           i32x4;

#define HH      64
#define WW      64
#define HW      4096
#define C_IN    512
#define K_REAL  537
#define K_TOT   544      // 512 (x) + 32 (gs padded)
#define K_PAD   32       // gs channels padded 25 -> 32
#define OUT_C   512
#define BM      128
#define BN      128
#define KC      32
#define KCP     40       // padded LDS row stride (bf16 elems); 80B keeps 16B align
#define NCHUNK  17       // 544 / 32

#if __has_builtin(__builtin_amdgcn_tensor_load_to_lds)
#define ATH_HAVE_TDM 1
#else
#define ATH_HAVE_TDM 0
#endif
#if __has_include(<hip/amd_detail/amd_gfx1250_TDM.h>)
#define ATH_TDM6 1
#else
#define ATH_TDM6 0
#endif

__device__ __forceinline__ unsigned short f2bf(float f) {
    unsigned int u = __float_as_uint(f);
    u += 0x7FFFu + ((u >> 16) & 1u);   // round-to-nearest-even
    return (unsigned short)(u >> 16);
}

// ---------------------------------------------------------------------------
// Phase 1a: correlation volume + L2 norm -> gs_bf (b, 32, h, w) bf16
// ---------------------------------------------------------------------------
__global__ __launch_bounds__(256) void corr_kernel(const float* __restrict__ x,
                                                   unsigned short* __restrict__ gs_bf) {
    const int tx = threadIdx.x & 15;
    const int ty = threadIdx.x >> 4;
    const int w0 = blockIdx.x * 16;
    const int h0 = blockIdx.y * 16;
    const int b  = blockIdx.z;

    __shared__ float4 tile[20][20];      // 16x16 inner + 2-halo, 4 channels deep

    float acc[25];
#pragma unroll
    for (int i = 0; i < 25; i++) acc[i] = 0.f;

    const float* xb = x + (size_t)b * C_IN * HW;

    for (int cc = 0; cc < C_IN; cc += 4) {
        for (int lid = threadIdx.x; lid < 400; lid += 256) {
            int yy = lid / 20, xx = lid % 20;
            int h = h0 - 2 + yy, w = w0 - 2 + xx;
            float4 v = make_float4(0.f, 0.f, 0.f, 0.f);
            if (h >= 0 && h < HH && w >= 0 && w < WW) {
                const float* p = xb + (size_t)cc * HW + h * WW + w;
                v.x = p[0]; v.y = p[HW]; v.z = p[2 * HW]; v.w = p[3 * HW];
            }
            tile[yy][xx] = v;
        }
        __syncthreads();
        float4 c = tile[ty + 2][tx + 2];
#pragma unroll
        for (int dy = 0; dy < 5; dy++)
#pragma unroll
            for (int dx = 0; dx < 5; dx++) {
                float4 nb = tile[ty + dy][tx + dx];
                acc[dy * 5 + dx] += c.x * nb.x + c.y * nb.y + c.z * nb.z + c.w * nb.w;
            }
        __syncthreads();
    }

    float ss = 0.f;
#pragma unroll
    for (int i = 0; i < 25; i++) ss += acc[i] * acc[i];
    float inv = 1.0f / sqrtf(ss + 1e-6f);

    int pix = (h0 + ty) * WW + (w0 + tx);
    unsigned short* g = gs_bf + (size_t)b * K_PAD * HW + pix;
#pragma unroll
    for (int i = 0; i < 25; i++) g[(size_t)i * HW] = f2bf(acc[i] * inv);
#pragma unroll
    for (int i = 25; i < 32; i++) g[(size_t)i * HW] = 0;
}

// ---------------------------------------------------------------------------
// Phase 1b: W fp32 (512 x 537) -> bf16 (512 x 544, zero padded K)
// ---------------------------------------------------------------------------
__global__ __launch_bounds__(256) void wconv_kernel(const float* __restrict__ Wf,
                                                    unsigned short* __restrict__ Wbf) {
    int idx = blockIdx.x * 256 + threadIdx.x;
    if (idx >= OUT_C * K_TOT) return;
    int o = idx / K_TOT, k = idx % K_TOT;
    float v = (k < K_REAL) ? Wf[(size_t)o * K_REAL + k] : 0.f;
    Wbf[idx] = f2bf(v);
}

// ---------------------------------------------------------------------------
// Phase 2: out[512, 65536] = relu(W[512,544] x feat[544, 65536] + bias)
// bf16 WMMA, 128x128 block tile, double-buffered LDS, K chunks of 32.
// A tile staged by the Tensor Data Mover (one wave issues tensor_load_to_lds,
// TDM pad feature reproduces the 40-element bank-stagger row stride).
// ---------------------------------------------------------------------------
__global__ __launch_bounds__(256) void gemm_kernel(const float* __restrict__ x,
                                                   const unsigned short* __restrict__ Wbf,
                                                   const unsigned short* __restrict__ gs_bf,
                                                   const float* __restrict__ bias,
                                                   float* __restrict__ out) {
    __shared__ unsigned short As[2][BM * KCP];   // W tile,   [m][k] k-contiguous
    __shared__ unsigned short Bs[2][BN * KCP];   // feat tile,[n][k] k-contiguous

    const int t    = threadIdx.x;
    const int lane = t & 31;
    const int wid  = t >> 5;
    const int wm   = wid >> 2;          // 0..1 -> 64-row sub-tile
    const int wn   = wid & 3;           // 0..3 -> 32-col sub-tile
    const int hi   = lane >> 4;
    const int l15  = lane & 15;

    const int b     = blockIdx.y;
    const int s0    = blockIdx.x * BN;          // pixel base inside batch plane
    const int oBase = blockIdx.z * BM;          // out-channel base

    const float*          xb = x     + (size_t)b * C_IN  * HW + s0;
    const unsigned short* gb = gs_bf + (size_t)b * K_PAD * HW + s0;

    f32x8 acc[4][2];
#pragma unroll
    for (int mf = 0; mf < 4; mf++)
#pragma unroll
        for (int nf = 0; nf < 2; nf++) acc[mf][nf] = (f32x8)(0.f);

#if ATH_HAVE_TDM
    // TDM 2D descriptor load: 32 (contig k) x 128 (rows) bf16 tile of Wbf,
    // row stride 544; LDS pad: every 16 DWORDs (one row) insert 4 DWORDs
    // -> LDS row stride = 40 bf16 = KCP.
    auto loadA_tdm = [&](int buf, int kc) {
        const unsigned short* src = Wbf + (size_t)oBase * K_TOT + kc * KC;
        unsigned long long ga = (unsigned long long)(uintptr_t)src;
        unsigned la = (unsigned)(uintptr_t)(&As[buf][0]);
        u32x4 g0;
        g0.x = 1u;                                          // count=1, user D#
        g0.y = la;                                          // lds_addr
        g0.z = (unsigned)ga;                                // global_addr[31:0]
        g0.w = (unsigned)((ga >> 32) & 0x01FFFFFFull) | (2u << 30);  // ga[56:32] | type=2
        i32x8 g1;
        g1[0] = (int)((1u << 16)      // data_size = 2B
                    | (1u << 20)      // pad_enable
                    | (3u << 22)      // pad_interval: 16 DWORDs (enc 3)
                    | (3u << 25));    // pad_amount: 4 DWORDs (enc 3)
        g1[1] = (int)((unsigned)(K_TOT & 0xFFFF) << 16);    // tensor_dim0 = 544
        g1[2] = (int)((unsigned)(OUT_C & 0xFFFF) << 16);    // tensor_dim1 = 512
        g1[3] = (int)((unsigned)(KC & 0xFFFF) << 16);       // tile_dim0 = 32
        g1[4] = BM;                                         // tile_dim1 = 128, tile_dim2 = 0
        g1[5] = K_TOT;                                      // tensor_dim0_stride = 544
        g1[6] = 0;
        g1[7] = 0;
        i32x4 gz = {0, 0, 0, 0};
#if ATH_TDM6
        i32x8 z8 = {0, 0, 0, 0, 0, 0, 0, 0};
        __builtin_amdgcn_tensor_load_to_lds(g0, g1, gz, gz, z8, 0);
#else
        __builtin_amdgcn_tensor_load_to_lds(g0, g1, gz, gz, 0);
#endif
    };
#else
    auto loadA = [&](int buf, int kc) {
        const unsigned short* src = Wbf + (size_t)oBase * K_TOT + kc * KC;
#pragma unroll
        for (int i = 0; i < 2; i++) {
            int lid = t + i * 256;                 // 0..511
            int row = lid >> 2;                    // 0..127
            int ko  = (lid & 3) * 8;               // 0,8,16,24
            u32x4 v = *(const u32x4*)(src + (size_t)row * K_TOT + ko);
            *(u32x4*)(&As[buf][row * KCP + ko]) = v;
        }
    };
#endif

    auto loadB = [&](int buf, int kc) {
        int k0 = kc * KC;
        if (k0 < C_IN) {                           // x part: fp32 -> bf16 on the fly
#pragma unroll
            for (int i = 0; i < 4; i++) {
                int lid = t + i * 256;             // 0..1023
                int n   = lid & 127;
                int kq  = (lid >> 7) * 4;          // 0,4,...,28
                const float* p = xb + (size_t)(k0 + kq) * HW + n;
                u32x2 pk;
                pk.x = (unsigned)f2bf(p[0])      | ((unsigned)f2bf(p[HW])     << 16);
                pk.y = (unsigned)f2bf(p[2 * HW]) | ((unsigned)f2bf(p[3 * HW]) << 16);
                *(u32x2*)(&Bs[buf][n * KCP + kq]) = pk;
            }
        } else {                                   // gs part: already bf16
#pragma unroll
            for (int i = 0; i < 4; i++) {
                int lid = t + i * 256;
                int n   = lid & 127;
                int kq  = (lid >> 7) * 4;
                const unsigned short* p = gb + (size_t)kq * HW + n;
                u32x2 pk;
                pk.x = (unsigned)p[0]      | ((unsigned)p[HW]     << 16);
                pk.y = (unsigned)p[2 * HW] | ((unsigned)p[3 * HW] << 16);
                *(u32x2*)(&Bs[buf][n * KCP + kq]) = pk;
            }
        }
    };

    union Frag { bf16x16 v; u32x4 q[2]; };

    auto compute = [&](int buf) {
        const unsigned short* A = As[buf];
        const unsigned short* B = Bs[buf];
        Frag a[4], bf[2];
        // A 16x32 bf16 layout: lanes<16 hold K0-7 & K16-23; lanes>=16 hold K8-15 & K24-31
#pragma unroll
        for (int mf = 0; mf < 4; mf++) {
            int m = wm * 64 + mf * 16 + l15;
            a[mf].q[0] = *(const u32x4*)(A + m * KCP + hi * 8);
            a[mf].q[1] = *(const u32x4*)(A + m * KCP + 16 + hi * 8);
        }
        // B 32x16 bf16 layout: lanes<16 hold K0-7 & K16-23 of col n; lanes>=16 hold K8-15 & K24-31
#pragma unroll
        for (int nf = 0; nf < 2; nf++) {
            int n = wn * 32 + nf * 16 + l15;
            bf[nf].q[0] = *(const u32x4*)(B + n * KCP + hi * 8);
            bf[nf].q[1] = *(const u32x4*)(B + n * KCP + 16 + hi * 8);
        }
#pragma unroll
        for (int mf = 0; mf < 4; mf++)
#pragma unroll
            for (int nf = 0; nf < 2; nf++)
                acc[mf][nf] = __builtin_amdgcn_wmma_f32_16x16x32_bf16(
                    false, a[mf].v, false, bf[nf].v,
                    (short)0, acc[mf][nf], false, false);
    };

#if ATH_HAVE_TDM
    const bool tdmWave = (wid == 0);
    if (tdmWave) loadA_tdm(0, 0);
    loadB(0, 0);
    if (tdmWave) __builtin_amdgcn_s_wait_tensorcnt(0);
    __syncthreads();
#pragma unroll 1
    for (int kc = 0; kc < NCHUNK; kc++) {
        int cur = kc & 1;
        if (kc + 1 < NCHUNK) {
            if (tdmWave) loadA_tdm(cur ^ 1, kc + 1);   // async DMA overlaps compute
            loadB(cur ^ 1, kc + 1);
        }
        compute(cur);
        if (tdmWave && (kc + 1 < NCHUNK)) __builtin_amdgcn_s_wait_tensorcnt(0);
        __syncthreads();
    }
#else
    loadA(0, 0);
    loadB(0, 0);
    __syncthreads();
#pragma unroll 1
    for (int kc = 0; kc < NCHUNK; kc++) {
        int cur = kc & 1;
        if (kc + 1 < NCHUNK) { loadA(cur ^ 1, kc + 1); loadB(cur ^ 1, kc + 1); }
        compute(cur);
        __syncthreads();
    }
#endif

    // Epilogue: C frag layout: VGPR r, lanes<16 -> M=r, lanes>=16 -> M=r+8; N=lane&15
    float* ob = out + (size_t)b * OUT_C * HW + s0;
#pragma unroll
    for (int mf = 0; mf < 4; mf++) {
#pragma unroll
        for (int r = 0; r < 8; r++) {
            int o  = oBase + wm * 64 + mf * 16 + hi * 8 + r;
            float bv = bias[o];
#pragma unroll
            for (int nf = 0; nf < 2; nf++) {
                int n = wn * 32 + nf * 16 + l15;
                float v = acc[mf][nf][r] + bv;
                v = v > 0.f ? v : 0.f;
                ob[(size_t)o * HW + n] = v;
            }
        }
    }
}

// ---------------------------------------------------------------------------
extern "C" void kernel_launch(void* const* d_in, const int* in_sizes, int n_in,
                              void* d_out, int out_size, void* d_ws, size_t ws_size,
                              hipStream_t stream) {
    const float* x    = (const float*)d_in[0];   // (16,512,64,64)
    const float* Wf   = (const float*)d_in[1];   // (512,537)
    const float* bias = (const float*)d_in[2];   // (512,)
    float* out        = (float*)d_out;           // (16,512,64,64)

    unsigned short* Wbf   = (unsigned short*)d_ws;                       // 512*544*2 B
    unsigned short* gs_bf = (unsigned short*)((char*)d_ws + (1 << 20));  // 16*32*4096*2 B

    wconv_kernel<<<(OUT_C * K_TOT + 255) / 256, 256, 0, stream>>>(Wf, Wbf);

    dim3 g1(WW / 16, HH / 16, 16);
    corr_kernel<<<g1, 256, 0, stream>>>(x, gs_bf);

    dim3 g2(HW / BN, 16, OUT_C / BM);   // (32, 16, 4)
    gemm_kernel<<<g2, 256, 0, stream>>>(x, Wbf, gs_bf, bias, out);
}
